// Scaled_Dot_Product_Attention_15599321219269
// MI455X (gfx1250) — matrix operand order repaired and checked by reading, not compiled, not run on hardware
//
#include <hip/hip_runtime.h>
#include <stdint.h>

// ---------------------------------------------------------------------------
// Flash-attention forward, B=2,H=16,S=2048,D=128 (fp32 in/out, bool mask).
// CDNA5 / gfx1250 wave32, v_wmma_f32_16x16x32_bf16.
//  * scores computed transposed (S^T = K Q^T): softmax reductions in-lane +
//    one shfl_xor(16); mask = two 8B loads/tile; S^T C-layout == P A-layout.
//  * double-buffered K/V LDS tiles -> ONE barrier per tile, stage/compute
//    overlap across waves.
//  * O rescale + corr broadcasts skipped via wave-uniform ballot when the
//    running max did not change (common after the first few tiles).
// ---------------------------------------------------------------------------

typedef __attribute__((ext_vector_type(16))) __bf16 v16bf;
typedef __attribute__((ext_vector_type(8)))  __bf16 v8bf;
typedef __attribute__((ext_vector_type(8)))  float  v8f;

#define ATT_B 2
#define ATT_H 16
#define ATT_S 2048
#define ATT_D 128

#define NWAVE 8
#define NTHR  (NWAVE * 32)
#define BM    (NWAVE * 16)   // 128 query rows per block
#define BN    32             // keys per tile
#define KPAD  136            // K lds row stride, bf16 (68 dwords, conflict-free reads)
#define VPAD  40             // V^T lds row stride, bf16 (20 dwords, conflict-free reads)

static __device__ __forceinline__ v16bf ld16bf(const __bf16* p0, const __bf16* p1) {
    v8bf lo = *(const v8bf*)p0;
    v8bf hi = *(const v8bf*)p1;
    v16bf r;
#pragma unroll
    for (int i = 0; i < 8; ++i) { r[i] = lo[i]; r[i + 8] = hi[i]; }
    return r;
}

__global__ __launch_bounds__(NTHR)
void Scaled_Dot_Product_Attention_15599321219269_kernel(
        const float* __restrict__ Q, const float* __restrict__ K,
        const float* __restrict__ V, const unsigned char* __restrict__ mask,
        float* __restrict__ Out) {

    __shared__ __bf16 Klds[2][BN][KPAD];       // [buf][key][d]  row-major
    __shared__ __bf16 VldsT[2][ATT_D][VPAD];   // [buf][d][key]  transposed

    const int tid  = threadIdx.x;
    const int lane = tid & 31;
    const int wave = tid >> 5;
    const int ln   = lane & 15;             // query column in S^T frags
    const int hs   = (lane >> 4) & 1;       // lane-half select

    const int qtile = blockIdx.x;
    const int bh    = blockIdx.y;
    const int qbase = qtile * BM + wave * 16;
    const size_t base = (size_t)bh * ATT_S * ATT_D;

    const float NEG_INF = -__builtin_inff();
    // fold 1/sqrt(128) * log2(e): softmax in exp2 domain
    const float QS = 0.08838834764831845f * 1.4426950408889634f;

    // ---- Q as B-fragments of S^T = K Q^T ----------------------------------
    // B 32x16 layout: lane col = ln, slot e -> kdim = e + 16hs -> contiguous d.
    v16bf qf[4];
    {
        const float* qrow = Q + base + (size_t)(qbase + ln) * ATT_D;
#pragma unroll
        for (int c = 0; c < 4; ++c) {
            float t[16];
            const float* g = qrow + 32 * c + 16 * hs;
            *(float4*)(t + 0)  = *(const float4*)(g + 0);
            *(float4*)(t + 4)  = *(const float4*)(g + 4);
            *(float4*)(t + 8)  = *(const float4*)(g + 8);
            *(float4*)(t + 12) = *(const float4*)(g + 12);
#pragma unroll
            for (int i = 0; i < 16; ++i) qf[c][i] = (__bf16)t[i];
        }
    }

    // ---- accumulators ------------------------------------------------------
    v8f of[8];
#pragma unroll
    for (int f = 0; f < 8; ++f) of[f] = (v8f){0.f,0.f,0.f,0.f,0.f,0.f,0.f,0.f};
    float m_run = NEG_INF;    // running stats for query qbase + ln
    float l_run = 0.f;

    const unsigned char* mq = mask + (size_t)(qbase + ln) * ATT_S;

    const int NT = ATT_S / BN;
    for (int kt = 0; kt < NT; ++kt) {
        const int kbase = kt * BN;
        const int buf   = kt & 1;

        // ---- stage K (row-major) / V (transposed) as bf16 -----------------
        // tile = 32 keys x 128 d = 1024 float4; 4 per thread, coalesced.
#pragma unroll
        for (int j = 0; j < 4; ++j) {
            const int idx = j * NTHR + tid;
            const int key = idx >> 5;
            const int d4  = idx & 31;
            const size_t goff = base + (size_t)(kbase + key) * ATT_D + d4 * 4;
            float4 kv = *(const float4*)(K + goff);
            __bf16* kd = &Klds[buf][key][d4 * 4];
            kd[0] = (__bf16)kv.x; kd[1] = (__bf16)kv.y;
            kd[2] = (__bf16)kv.z; kd[3] = (__bf16)kv.w;
            float4 vv = *(const float4*)(V + goff);
            VldsT[buf][d4 * 4 + 0][key] = (__bf16)vv.x;
            VldsT[buf][d4 * 4 + 1][key] = (__bf16)vv.y;
            VldsT[buf][d4 * 4 + 2][key] = (__bf16)vv.z;
            VldsT[buf][d4 * 4 + 3][key] = (__bf16)vv.w;
        }
        if (kt + 1 < NT) {  // speculative prefetch of the next tile
            const size_t noff = base + (size_t)(kbase + BN) * ATT_D + (size_t)tid * 16;
            __builtin_prefetch(K + noff, 0, 1);
            __builtin_prefetch(V + noff, 0, 1);
        }
        __syncthreads();   // single barrier: double-buffer makes the trailing
                           // barrier unnecessary (see header comment)

        // ---- S^T = K Q^T : frag0 = keys [0,16), frag1 = keys [16,32) ------
        v8f s0 = (v8f){0.f,0.f,0.f,0.f,0.f,0.f,0.f,0.f};
        v8f s1 = s0;
#pragma unroll
        for (int c = 0; c < 4; ++c) {
            const __bf16* ka0 = &Klds[buf][ln][32 * c + 16 * hs];
            const __bf16* ka1 = &Klds[buf][16 + ln][32 * c + 16 * hs];
            v16bf k0 = ld16bf(ka0, ka0 + 8);    // A-frag groups at +0 / +8 (contiguous k)
            v16bf k1 = ld16bf(ka1, ka1 + 8);
            s0 = __builtin_amdgcn_wmma_f32_16x16x32_bf16(
                     false, k0, false, qf[c], (short)0, s0, false, false);
            s1 = __builtin_amdgcn_wmma_f32_16x16x32_bf16(
                     false, k1, false, qf[c], (short)0, s1, false, false);
        }

        // ---- masked online softmax ----------------------------------------
        const uint64_t mm0 = *(const uint64_t*)(mq + kbase + 8 * hs);
        const uint64_t mm1 = *(const uint64_t*)(mq + kbase + 16 + 8 * hs);

        float tmax = NEG_INF;
#pragma unroll
        for (int v = 0; v < 8; ++v) {
            float a = s0[v] * QS;
            float b = s1[v] * QS;
            if (!((mm0 >> (8 * v)) & 0xff)) a = NEG_INF;
            if (!((mm1 >> (8 * v)) & 0xff)) b = NEG_INF;
            s0[v] = a; s1[v] = b;
            tmax = fmaxf(tmax, fmaxf(a, b));
        }
        tmax = fmaxf(tmax, __shfl_xor(tmax, 16, 32));

        const float mn   = fmaxf(m_run, tmax);
        const bool  dead = (mn == NEG_INF);
        const bool  upd  = (mn > m_run);           // running max changed?
        const float corr = upd ? exp2f(m_run - mn) : 1.0f;

        // S^T C-layout == P A-layout: pack probabilities straight into pa.
        v16bf pa;
        float rs = 0.f;
#pragma unroll
        for (int v = 0; v < 8; ++v) {
            float p0 = dead ? 0.0f : exp2f(s0[v] - mn);
            float p1 = dead ? 0.0f : exp2f(s1[v] - mn);
            pa[v]     = (__bf16)p0;
            pa[v + 8] = (__bf16)p1;
            rs += p0 + p1;
        }
        rs += __shfl_xor(rs, 16, 32);
        l_run = l_run * corr + rs;
        m_run = mn;

        // ---- rescale O only when some query's max changed (wave-uniform) --
        if (__ballot(upd)) {
            float corrv[8];
#pragma unroll
            for (int v = 0; v < 8; ++v)
                corrv[v] = __shfl(corr, v + 8 * hs, 32);
#pragma unroll
            for (int f = 0; f < 8; ++f) {
#pragma unroll
                for (int v = 0; v < 8; ++v) of[f][v] *= corrv[v];
            }
        }

        // ---- O += P V  (8 d-frags, K=32 over keys) ------------------------
#pragma unroll
        for (int f = 0; f < 8; ++f) {
            const __bf16* vp = &VldsT[buf][16 * f + ln][8 * hs];
            v16bf vb = ld16bf(vp, vp + 16);
            of[f] = __builtin_amdgcn_wmma_f32_16x16x32_bf16(
                        false, pa, false, vb, (short)0, of[f], false, false);
        }
    }

    // ---- epilogue: divide rows by their sums, store fp32 -------------------
    float inv[8];
#pragma unroll
    for (int v = 0; v < 8; ++v) {
        float lv = __shfl(l_run, v + 8 * hs, 32);
        inv[v] = 1.0f / lv;
    }
#pragma unroll
    for (int f = 0; f < 8; ++f) {
#pragma unroll
        for (int v = 0; v < 8; ++v) {
            Out[base + (size_t)(qbase + v + 8 * hs) * ATT_D + 16 * f + ln] =
                of[f][v] * inv[v];
        }
    }
}

extern "C" void kernel_launch(void* const* d_in, const int* in_sizes, int n_in,
                              void* d_out, int out_size, void* d_ws, size_t ws_size,
                              hipStream_t stream) {
    const float* Q = (const float*)d_in[0];
    const float* K = (const float*)d_in[1];
    const float* V = (const float*)d_in[2];
    const unsigned char* mask = (const unsigned char*)d_in[3];
    float* Out = (float*)d_out;

    dim3 grid(ATT_S / BM, ATT_B * ATT_H);   // (16, 32)
    dim3 block(NTHR);                       // 256 threads = 8 waves
    Scaled_Dot_Product_Attention_15599321219269_kernel
        <<<grid, block, 0, stream>>>(Q, K, V, mask, Out);
}
